// KANPredictor_14328010900291
// MI455X (gfx1250) — compile-verified, matrix operand order checked by reading, not run on hardware
//
#include <hip/hip_runtime.h>
#include <cstdint>
#include <cstddef>

// ---------------------------------------------------------------------------
// KAN predictor for MI455X (gfx1250, wave32, WMMA).
//   layer out = [silu(x) | T0..T12(tanh x)] @ [base_w; poly_w[0..12]] (bf16
//   WMMA GEMM, fp32 accum), then fp32 RMSNorm (decoupled pass).
// GEMM: 128x128 block tiles, 8 waves (each 16 rows x 128 cols), B staged in
// LDS in two 7-basis phases (56KB) for 8x M-reuse of the L2-resident weights.
// B staging uses GLOBAL_LOAD_ASYNC_TO_LDS (ASYNCcnt) when the toolchain
// exposes the builtin; otherwise falls back to load + ds_store.
// ---------------------------------------------------------------------------

typedef __attribute__((ext_vector_type(16))) __bf16 v16bf;
typedef __attribute__((ext_vector_type(8)))  float  v8f;
typedef int v4i_vs __attribute__((vector_size(4 * sizeof(int))));

#define D_MODEL 1024
#define HIDDEN  2048
#define Z_DIM   32
#define IN0     2080      // 2*D_MODEL + Z_DIM
#define N_BASIS 14        // 13 Chebyshev + silu/base
#define M_ROWS  16384     // B * N_tgt
#define N_TGT   2048
#define S_CTX   128
#define RMS_EPS 1e-6f
#define BLK_M   128
#define BLK_N   128

#if __has_builtin(__builtin_amdgcn_global_load_async_to_lds_b128)
#define HAVE_ASYNC_LDS 1
#else
#define HAVE_ASYNC_LDS 0
#endif

static __device__ __forceinline__ void cp_b128_to_lds(uint32_t* dst_lds,
                                                      const uint32_t* src) {
#if HAVE_ASYNC_LDS
  __builtin_amdgcn_global_load_async_to_lds_b128(
      (__attribute__((address_space(1))) v4i_vs*)src,
      (__attribute__((address_space(3))) v4i_vs*)dst_lds,
      /*offset=*/0, /*cpol=*/0);
#else
  *(uint4*)dst_lds = *(const uint4*)src;
#endif
}

static __device__ __forceinline__ void wait_async_copies() {
#if HAVE_ASYNC_LDS
#if __has_builtin(__builtin_amdgcn_s_wait_asynccnt)
  __builtin_amdgcn_s_wait_asynccnt(0);
#else
  asm volatile("s_wait_asynccnt 0x0" ::: "memory");
#endif
#endif
}

static __device__ __forceinline__ unsigned short f2bf(float f) {
  union { __bf16 h; unsigned short u; } cv; cv.h = (__bf16)f; return cv.u;
}

static __device__ __forceinline__ v16bf ldfrag(const uint32_t* p) {
  union { uint4 q[2]; v16bf v; } u;
  u.q[0] = *(const uint4*)(p);
  u.q[1] = *(const uint4*)(p + 4);
  return u.v;
}

static __device__ __forceinline__ v8f wmma_bf16(v16bf a, v16bf b, v8f c) {
  return __builtin_amdgcn_wmma_f32_16x16x32_bf16(
      false, a, false, b, (short)0, c, false, false);
}

// ---------------------------------------------------------------------------
// Pack fp32 weights -> bf16 fragments in the ISA 32x16 B-operand layout
// (§7.12.2): lane = N%16 (+16 selects K octets 8..15/24..31), VGPR halves
// hold K pairs.  Tile T = ((k*CB + cb)*NB + nb): 256 dwords, addr = T*256 +
// lane*8 + d.
// ---------------------------------------------------------------------------
__global__ void shuffle_weights(const float* __restrict__ base_w,
                                const float* __restrict__ poly_w,
                                uint32_t* __restrict__ Wf,
                                int in_dim, int out_dim)
{
  const int NB = out_dim >> 4;
  const int CB = in_dim >> 5;
  const size_t total = (size_t)N_BASIS * CB * NB * 256;
  for (size_t idx = (size_t)blockIdx.x * blockDim.x + threadIdx.x;
       idx < total; idx += (size_t)gridDim.x * blockDim.x) {
    const int d    = (int)(idx & 7);
    const int lane = (int)((idx >> 3) & 31);
    const size_t T = idx >> 8;
    const int nb   = (int)(T % NB);
    const size_t t2 = T / NB;
    const int cb   = (int)(t2 % CB);
    const int k    = (int)(t2 / CB);
    const int n    = (nb << 4) + (lane & 15);
    uint32_t outw = 0;
#pragma unroll
    for (int h = 0; h < 2; ++h) {
      const int e = 2 * d + h;
      const int K = (lane < 16) ? ((e < 8) ? e : e + 8)
                                : ((e < 8) ? e + 8 : e + 16);
      const int c = (cb << 5) + K;
      const float f = (k < 13)
          ? poly_w[((size_t)k * in_dim + c) * out_dim + n]
          : base_w[(size_t)c * out_dim + n];
      outw |= (uint32_t)f2bf(f) << (16 * h);
    }
    Wf[idx] = outw;
  }
}

// ---------------------------------------------------------------------------
// Layer-0 input rows: x = [h_x[b,last,:], pos_embed[tp], z]; emit
// silu(bf16(x)) as bf16 and tanh(x) as fp32.
// ---------------------------------------------------------------------------
__global__ void prep0(const float* __restrict__ h_x,
                      const int* __restrict__ tpos,
                      const float* __restrict__ z,
                      const float* __restrict__ pe,
                      unsigned short* __restrict__ s0,
                      float* __restrict__ t0)
{
  const int row = blockIdx.x;
  const int b   = row >> 11;
  const int tt  = row & (N_TGT - 1);
  const int p   = tpos[(size_t)b * N_TGT + tt];
  const float* ctx = h_x + ((size_t)b * S_CTX + (S_CTX - 1)) * D_MODEL;
  const float* pr  = pe + (size_t)p * D_MODEL;
  const float* zr  = z + ((size_t)b * N_TGT + tt) * Z_DIM;
  for (int c = threadIdx.x; c < IN0; c += blockDim.x) {
    const float x = (c < D_MODEL) ? ctx[c]
                  : (c < 2 * D_MODEL) ? pr[c - D_MODEL]
                  : zr[c - 2 * D_MODEL];
    const float xb  = (float)(__bf16)x;
    const float sig = 1.0f / (1.0f + __expf(-xb));
    s0[(size_t)row * IN0 + c] = f2bf(xb * sig);
    t0[(size_t)row * IN0 + c] = tanhf(x);
  }
}

__global__ void prep_act(const float* __restrict__ x,
                         unsigned short* __restrict__ s,
                         float* __restrict__ t, size_t n)
{
  for (size_t i = (size_t)blockIdx.x * blockDim.x + threadIdx.x;
       i < n; i += (size_t)gridDim.x * blockDim.x) {
    const float v   = x[i];
    const float vb  = (float)(__bf16)v;
    const float sig = 1.0f / (1.0f + __expf(-vb));
    s[i] = f2bf(vb * sig);
    t[i] = tanhf(v);
  }
}

__global__ void zero_f32(float* __restrict__ p, size_t n)
{
  for (size_t i = (size_t)blockIdx.x * blockDim.x + threadIdx.x;
       i < n; i += (size_t)gridDim.x * blockDim.x) p[i] = 0.0f;
}

// ---------------------------------------------------------------------------
// KAN GEMM: block = 128 rows x 128 cols, 8 waves (wave w: rows w*16..+16,
// all 128 cols -> 8 f32 accumulators).  K loop over in/32 blocks; per block
// the 14 basis B-tiles (32x128 bf16 each) are staged through LDS in two
// 7-basis phases; the Chebyshev recursion lives in fp32 registers and spans
// both phases.  Emits raw fp32 + per-row sumsq (global atomics).
// ---------------------------------------------------------------------------
__global__ void __launch_bounds__(256)
kan_gemm(const unsigned short* __restrict__ sact, // silu bf16, M x in
         const float* __restrict__ tact,          // tanh fp32, M x in
         const uint32_t* __restrict__ Wf,         // shuffled bf16 frags
         float* __restrict__ raw,                 // fp32 M x out (no norm)
         float* __restrict__ rowssq,              // fp32 M (pre-zeroed)
         int in_dim, int out_dim)
{
  const int lane = threadIdx.x & 31;
  const int wave = threadIdx.x >> 5;            // 0..7 = M subtile
  const int CB   = in_dim >> 5;
  const int NB   = out_dim >> 4;
  const int row0 = blockIdx.x * BLK_M + wave * 16;
  const int bn8  = blockIdx.y * (BLK_N >> 4);   // first 16-col tile of block
  const int m    = lane & 15;                   // A-operand: M = lane%16
  const int koff = (lane < 16) ? 0 : 8;         // K octet split

  __shared__ uint32_t Bs[7 * 2048];             // 7 basis tiles x 8 ntiles, 56KB

  const float* trow = tact + (size_t)(row0 + m) * in_dim;
  const unsigned short* srow = sact + (size_t)(row0 + m) * in_dim;

  v8f acc0 = {}, acc1 = {}, acc2 = {}, acc3 = {};
  v8f acc4 = {}, acc5 = {}, acc6 = {}, acc7 = {};

  v16bf ones;
#pragma unroll
  for (int i = 0; i < 16; ++i) ones[i] = (__bf16)1.0f;

#define MMK(AF, SLOT) do {                                                   \
    const uint32_t* bp = Bs + (SLOT) * 2048 + (lane << 3);                   \
    acc0 = wmma_bf16((AF), ldfrag(bp       ), acc0);                         \
    acc1 = wmma_bf16((AF), ldfrag(bp +  256), acc1);                         \
    acc2 = wmma_bf16((AF), ldfrag(bp +  512), acc2);                         \
    acc3 = wmma_bf16((AF), ldfrag(bp +  768), acc3);                         \
    acc4 = wmma_bf16((AF), ldfrag(bp + 1024), acc4);                         \
    acc5 = wmma_bf16((AF), ldfrag(bp + 1280), acc5);                         \
    acc6 = wmma_bf16((AF), ldfrag(bp + 1536), acc6);                         \
    acc7 = wmma_bf16((AF), ldfrag(bp + 1792), acc7);                         \
  } while (0)

#define STAGE(KIDX, SLOT) do {                                               \
    const uint32_t* src = Wf + (((size_t)(KIDX) * CB + cb) * NB + bn8) * 256;\
    uint32_t* dst = Bs + (SLOT) * 2048;                                      \
    cp_b128_to_lds(dst + (threadIdx.x << 2), src + (threadIdx.x << 2));      \
    cp_b128_to_lds(dst + ((threadIdx.x + 256) << 2),                         \
                   src + ((threadIdx.x + 256) << 2));                        \
  } while (0)

  for (int cb = 0; cb < CB; ++cb) {
    const int c0 = (cb << 5) + koff;
    // ---- A operands for this K block ----
    float4 xq0 = *(const float4*)(trow + c0);
    float4 xq1 = *(const float4*)(trow + c0 + 4);
    float4 xq2 = *(const float4*)(trow + c0 + 16);
    float4 xq3 = *(const float4*)(trow + c0 + 20);
    float xv[16] = {xq0.x, xq0.y, xq0.z, xq0.w, xq1.x, xq1.y, xq1.z, xq1.w,
                    xq2.x, xq2.y, xq2.z, xq2.w, xq3.x, xq3.y, xq3.z, xq3.w};
    union { uint4 q[2]; v16bf v; } su;
    su.q[0] = *(const uint4*)(srow + c0);
    su.q[1] = *(const uint4*)(srow + c0 + 16);

    // ---- phase 0: bases {silu, T0..T5} -> LDS slots 0..6 ----
    __syncthreads();
#pragma unroll
    for (int s = 0; s < 7; ++s) {
      const int k = (s == 0) ? 13 : (s - 1);
      STAGE(k, s);
    }
    wait_async_copies();
    __syncthreads();

    MMK(su.v, 0);                       // silu @ base_w
    MMK(ones, 1);                       // T0 = 1
    float tp[16], tc[16];
    v16bf af;
#pragma unroll
    for (int i = 0; i < 16; ++i) { tp[i] = 1.0f; tc[i] = xv[i]; af[i] = (__bf16)xv[i]; }
    MMK(af, 2);                         // T1 = tanh(x)
#pragma unroll
    for (int k = 2; k <= 5; ++k) {      // T2..T5 -> slots 3..6
#pragma unroll
      for (int i = 0; i < 16; ++i) {
        const float tn = 2.0f * xv[i] * tc[i] - tp[i];
        tp[i] = tc[i]; tc[i] = tn; af[i] = (__bf16)tn;
      }
      MMK(af, k + 1);
    }

    // ---- phase 1: bases T6..T12 -> LDS slots 0..6 ----
    __syncthreads();
#pragma unroll
    for (int s = 0; s < 7; ++s) STAGE(s + 6, s);
    wait_async_copies();
    __syncthreads();

#pragma unroll
    for (int k = 6; k <= 12; ++k) {
#pragma unroll
      for (int i = 0; i < 16; ++i) {
        const float tn = 2.0f * xv[i] * tc[i] - tp[i];
        tp[i] = tc[i]; tc[i] = tn; af[i] = (__bf16)tn;
      }
      MMK(af, k - 6);
    }
  }
#undef STAGE
#undef MMK

  // ---- per-row sum of squares (C/D layout: VGPR j -> row j + 8*(lane>=16),
  //      col n*16 + lane%16), reduced across the 16 lanes of a row ----
  float ss[8];
#pragma unroll
  for (int j = 0; j < 8; ++j)
    ss[j] = acc0[j]*acc0[j] + acc1[j]*acc1[j] + acc2[j]*acc2[j] + acc3[j]*acc3[j]
          + acc4[j]*acc4[j] + acc5[j]*acc5[j] + acc6[j]*acc6[j] + acc7[j]*acc7[j];
#pragma unroll
  for (int j = 0; j < 8; ++j) {
    ss[j] += __shfl_xor(ss[j], 1, 32);
    ss[j] += __shfl_xor(ss[j], 2, 32);
    ss[j] += __shfl_xor(ss[j], 4, 32);
    ss[j] += __shfl_xor(ss[j], 8, 32);
  }
  const int hi = lane >> 4;
  if ((lane & 15) == 0) {
#pragma unroll
    for (int j = 0; j < 8; ++j)
      atomicAdd(&rowssq[row0 + hi * 8 + j], ss[j]);
  }

#define STOREN(ACC, N) do {                                                  \
    const int col = ((bn8 + (N)) << 4) + (lane & 15);                        \
    _Pragma("unroll")                                                        \
    for (int j = 0; j < 8; ++j)                                              \
      raw[(size_t)(row0 + hi * 8 + j) * out_dim + col] = (ACC)[j];           \
  } while (0)
  STOREN(acc0, 0); STOREN(acc1, 1); STOREN(acc2, 2); STOREN(acc3, 3);
  STOREN(acc4, 4); STOREN(acc5, 5); STOREN(acc6, 6); STOREN(acc7, 7);
#undef STOREN
}

// Apply RMSNorm: out = raw * rsqrt(mean(raw^2) + eps) * rms_scale
__global__ void rmsnorm_apply(const float* __restrict__ raw,
                              const float* __restrict__ rowssq,
                              const float* __restrict__ rmsw,
                              float* __restrict__ out, int out_dim)
{
  const int row = blockIdx.x;
  const float sc = rsqrtf(rowssq[row] / (float)out_dim + RMS_EPS);
  const size_t base = (size_t)row * out_dim;
  for (int c = blockIdx.y * blockDim.x + threadIdx.x; c < out_dim;
       c += gridDim.y * blockDim.x)
    out[base + c] = raw[base + c] * sc * rmsw[c];
}

// ---------------------------------------------------------------------------
extern "C" void kernel_launch(void* const* d_in, const int* in_sizes, int n_in,
                              void* d_out, int out_size, void* d_ws, size_t ws_size,
                              hipStream_t stream)
{
  const float* h_x   = (const float*)d_in[0];
  const int*   tpos  = (const int*)  d_in[1];
  const float* z     = (const float*)d_in[2];
  const float* pe    = (const float*)d_in[3];
  const float* base0 = (const float*)d_in[4];
  const float* poly0 = (const float*)d_in[5];
  const float* rms0  = (const float*)d_in[6];
  const float* base1 = (const float*)d_in[7];
  const float* poly1 = (const float*)d_in[8];
  const float* rms1  = (const float*)d_in[9];
  float* outp = (float*)d_out;
  (void)in_sizes; (void)n_in; (void)out_size; (void)ws_size;

  char* w = (char*)d_ws;
  size_t off = 0;
  auto take = [&](size_t bytes) -> char* {
    char* p = w + off;
    off = (off + bytes + 255) & ~(size_t)255;
    return p;
  };
  uint32_t*       Wf0  = (uint32_t*)      take((size_t)N_BASIS * IN0    * HIDDEN  * 2);
  uint32_t*       Wf1  = (uint32_t*)      take((size_t)N_BASIS * HIDDEN * D_MODEL * 2);
  unsigned short* s0   = (unsigned short*)take((size_t)M_ROWS * IN0    * 2);
  float*          t0   = (float*)         take((size_t)M_ROWS * IN0    * 4);
  float*          raw0 = (float*)         take((size_t)M_ROWS * HIDDEN * 4);
  float*          h1   = (float*)         take((size_t)M_ROWS * HIDDEN * 4);
  unsigned short* s1   = (unsigned short*)take((size_t)M_ROWS * HIDDEN * 2);
  float*          t1   = (float*)         take((size_t)M_ROWS * HIDDEN * 4);
  float*          raw1 = (float*)         take((size_t)M_ROWS * D_MODEL * 4);
  float*          ssq0 = (float*)         take((size_t)M_ROWS * 4);
  float*          ssq1 = (float*)         take((size_t)M_ROWS * 4);

  // Pack weights into WMMA B-fragment layout (bf16).
  shuffle_weights<<<16384, 256, 0, stream>>>(base0, poly0, Wf0, IN0,    HIDDEN);
  shuffle_weights<<<16384, 256, 0, stream>>>(base1, poly1, Wf1, HIDDEN, D_MODEL);
  zero_f32<<<64, 256, 0, stream>>>(ssq0, (size_t)M_ROWS);
  zero_f32<<<64, 256, 0, stream>>>(ssq1, (size_t)M_ROWS);

  // Layer 0
  prep0<<<M_ROWS, 128, 0, stream>>>(h_x, tpos, z, pe, s0, t0);
  kan_gemm<<<dim3(M_ROWS / BLK_M, HIDDEN / BLK_N), 256, 0, stream>>>(
      s0, t0, Wf0, raw0, ssq0, IN0, HIDDEN);
  rmsnorm_apply<<<dim3(M_ROWS, 8), 256, 0, stream>>>(raw0, ssq0, rms0, h1, HIDDEN);

  // Layer 1
  prep_act<<<8192, 256, 0, stream>>>(h1, s1, t1, (size_t)M_ROWS * HIDDEN);
  kan_gemm<<<dim3(M_ROWS / BLK_M, D_MODEL / BLK_N), 256, 0, stream>>>(
      s1, t1, Wf1, raw1, ssq1, HIDDEN, D_MODEL);
  rmsnorm_apply<<<dim3(M_ROWS, 4), 256, 0, stream>>>(raw1, ssq1, rms1, outp, D_MODEL);
}
